// MultiHeadAttentionWithMaskBlock_9517647527997
// MI455X (gfx1250) — matrix operand order repaired and checked by reading, not compiled, not run on hardware
//
#include <hip/hip_runtime.h>

#define DIMN   1024
#define HEADS  4
#define DH     256
#define BATCH  4
#define SEQ    2048
#define MTOT   (BATCH*SEQ)   // 8192 rows for the projection GEMMs

typedef unsigned short u16;
typedef __bf16        bf16x16 __attribute__((ext_vector_type(16)));
typedef float         f32x8   __attribute__((ext_vector_type(8)));
typedef unsigned int  u32x4   __attribute__((ext_vector_type(4)));
typedef unsigned int  u32x8   __attribute__((ext_vector_type(8)));

static __device__ __forceinline__ u16 f2bf(float f) {
    unsigned u = __float_as_uint(f);
    u += 0x7fffu + ((u >> 16) & 1u);   // round-to-nearest-even
    return (u16)(u >> 16);
}

static __device__ __forceinline__ bf16x16 pack_bf(u32x4 a, u32x4 b) {
    u32x8 u = __builtin_shufflevector(a, b, 0, 1, 2, 3, 4, 5, 6, 7);
    return __builtin_bit_cast(bf16x16, u);
}

static __device__ __forceinline__ f32x8 wmma_bf16(bf16x16 a, bf16x16 b, f32x8 c) {
    return __builtin_amdgcn_wmma_f32_16x16x32_bf16(false, a, false, b, (short)0, c,
                                                   false, false);
}

// A-operand (16x32 bf16, M rows): lane<16 -> {K0..7, K16..23}, lane>=16 -> {K8..15, K24..31}
static __device__ __forceinline__ bf16x16 load_aop(const u16* row_base, int hl) {
    const u16* p = row_base + hl * 8;           // +16B for upper half-wave
    u32x4 x = *(const u32x4*)(p);
    u32x4 y = *(const u32x4*)(p + 16);          // +32 bytes
    return pack_bf(x, y);
}

// B-operand (32x16 bf16 = Bt rows of 32 K): lane<16 -> {K0..7, K16..23}, lane>=16 -> {K8..15, K24..31}
static __device__ __forceinline__ bf16x16 load_bop(const u16* row_base, int hl) {
    const u16* p = row_base + hl * 8;           // +16B for upper half-wave
    u32x4 x = *(const u32x4*)(p);
    u32x4 y = *(const u32x4*)(p + 16);          // +32 bytes
    return pack_bf(x, y);
}

// CDNA5 async DMA: 16B per lane, global -> LDS, tracked by ASYNCcnt (no VGPR data path)
static __device__ __forceinline__ void async_copy16(void* lds, const void* g) {
    asm volatile("global_load_async_to_lds_b128 %0, %1, off"
                 :: "v"((unsigned)(unsigned long long)(size_t)lds),
                    "v"((unsigned long long)(size_t)g)
                 : "memory");
}
static __device__ __forceinline__ void wait_async0() {
    asm volatile("s_wait_asynccnt 0x0" ::: "memory");
}

// ---------------------------------------------------------------- f32 -> bf16
__global__ void cvt_kernel(const float* __restrict__ in, u16* __restrict__ out, int n) {
    int i = blockIdx.x * 256 + threadIdx.x;
    if (i < n) out[i] = f2bf(in[i]);
}

// ------------------------------------------------------- GEMM: Y = X*W^T + b
// X: [MTOT][1024] bf16 row-major (A).  Wt: [1024][1024] bf16 row-major = B^T rows.
// MODE 0: bf16 out, split-head  [B,H,T,DH]
// MODE 1: bf16 out, transposed  [B,H,DH,T]   (for V^T)
// MODE 2: f32  out, plain       [MTOT][1024] (final projection -> d_out)
template <int MODE>
__global__ __launch_bounds__(256) void gemm_bf16_kernel(
        const u16* __restrict__ A, const u16* __restrict__ Wt,
        const float* __restrict__ bias, void* __restrict__ outp)
{
    __shared__ __align__(16) unsigned char ldsA[128 * 80];
    __shared__ __align__(16) unsigned char ldsB[128 * 80];

    const int tid = threadIdx.x, lane = tid & 31, w = tid >> 5;
    const int hl = lane >> 4, l15 = lane & 15;
    const int m0 = blockIdx.x * 128, n0 = blockIdx.y * 128;
    const int waveM = (w & 3) * 32, waveN = (w >> 2) * 64;

    f32x8 acc[2][4] = {};

    for (int kk = 0; kk < DIMN; kk += 32) {
        // async-stage tiles: 128 rows x 32 bf16 (64B) each, 80B padded LDS rows
#pragma unroll
        for (int t = 0; t < 2; ++t) {
            int ch = tid + t * 256;          // 0..511
            int row = ch >> 2, seg = ch & 3;
            async_copy16(ldsA + row * 80 + seg * 16,
                         A + (m0 + row) * DIMN + kk + seg * 8);
            async_copy16(ldsB + row * 80 + seg * 16,
                         Wt + (n0 + row) * DIMN + kk + seg * 8);
        }
        if (kk + 32 < DIMN) {   // global_prefetch_b8 of next K-slab
            __builtin_prefetch(A + (m0 + (tid >> 1)) * DIMN + kk + 32, 0, 1);
            __builtin_prefetch(Wt + (n0 + (tid >> 1)) * DIMN + kk + 32, 0, 1);
        }
        wait_async0();           // own async ops landed in LDS
        __syncthreads();         // everyone's ops landed

        bf16x16 aop[2], bop[4];
#pragma unroll
        for (int i = 0; i < 2; ++i)
            aop[i] = load_aop((const u16*)(ldsA + (waveM + i * 16 + l15) * 80), hl);
#pragma unroll
        for (int j = 0; j < 4; ++j)
            bop[j] = load_bop((const u16*)(ldsB + (waveN + j * 16 + l15) * 80), hl);
#pragma unroll
        for (int i = 0; i < 2; ++i)
#pragma unroll
            for (int j = 0; j < 4; ++j)
                acc[i][j] = wmma_bf16(aop[i], bop[j], acc[i][j]);
        __syncthreads();
    }

#pragma unroll
    for (int i = 0; i < 2; ++i) {
#pragma unroll
        for (int j = 0; j < 4; ++j) {
            int ncol = n0 + waveN + j * 16 + l15;
            float bi = bias[ncol];
#pragma unroll
            for (int e = 0; e < 8; ++e) {
                int R = m0 + waveM + i * 16 + hl * 8 + e;   // C layout: M = e + half*8
                float v = acc[i][j][e] + bi;
                if (MODE == 2) {
                    ((float*)outp)[(long)R * DIMN + ncol] = v;
                } else {
                    int b = R >> 11, t = R & (SEQ - 1);
                    int hh = ncol >> 8, d = ncol & (DH - 1);
                    u16* O = (u16*)outp;
                    if (MODE == 0)
                        O[((b * HEADS + hh) * SEQ + t) * DH + d] = f2bf(v);
                    else
                        O[((b * HEADS + hh) * DH + d) * SEQ + t] = f2bf(v);
                }
            }
        }
    }
}

// -------------------------- Phase A: per-column (per k) online max/sum stats
__global__ __launch_bounds__(256) void attn_stats_kernel(
        const u16* __restrict__ Qb, const u16* __restrict__ Kb,
        const int* __restrict__ mask,
        float* __restrict__ mstat, float* __restrict__ lstat)
{
    const int lane = threadIdx.x & 31, w = threadIdx.x >> 5;
    const int hl = lane >> 4, l15 = lane & 15;
    const int bh = blockIdx.y, b = bh >> 2;
    const int kw0 = blockIdx.x * 128 + w * 16;   // this wave's 16 k-columns
    const u16* Qh = Qb + bh * (SEQ * DH);
    const u16* Kh = Kb + bh * (SEQ * DH);
    const float scale = 0.0625f;                 // 1/sqrt(256)

    // resident B-operands: K rows kw0..kw0+15, full d=256 (8 chunks of K=32)
    bf16x16 bop[8];
#pragma unroll
    for (int c = 0; c < 8; ++c)
        bop[c] = load_bop(Kh + (kw0 + l15) * DH + c * 32, hl);

    float mr[8], lr[8];
#pragma unroll
    for (int e = 0; e < 8; ++e) { mr[e] = -3.0e38f; lr[e] = 0.f; }

    for (int qt = 0; qt < SEQ / 16; ++qt) {
        f32x8 s = {};
        const u16* qr = Qh + (qt * 16 + l15) * DH;
#pragma unroll
        for (int c = 0; c < 8; ++c)
            s = wmma_bf16(load_aop(qr + c * 32, hl), bop[c], s);

        const int kcol = kw0 + l15;
        const int qbase = qt * 16 + hl * 8;
#pragma unroll
        for (int e = 0; e < 8; ++e) {
            int mv = mask[(b * SEQ + qbase + e) * SEQ + kcol];
            float sv = mv ? s[e] * scale : -1.0e9f;
            float mn = fmaxf(mr[e], sv);
            lr[e] = lr[e] * __expf(mr[e] - mn) + __expf(sv - mn);
            mr[e] = mn;
        }
    }
    // reduce 8 rows held by this lane, then combine lane <-> lane^16 (same column)
    float mc = mr[0];
#pragma unroll
    for (int e = 1; e < 8; ++e) mc = fmaxf(mc, mr[e]);
    float lc = 0.f;
#pragma unroll
    for (int e = 0; e < 8; ++e) lc += lr[e] * __expf(mr[e] - mc);
    float mo = __shfl_xor(mc, 16, 32);
    float lo = __shfl_xor(lc, 16, 32);
    float M = fmaxf(mc, mo);
    float L = lc * __expf(mc - M) + lo * __expf(mo - M);
    if (lane < 16) {
        mstat[bh * SEQ + kw0 + l15] = M;
        lstat[bh * SEQ + kw0 + l15] = L;
    }
}

// ---------------- Phase B: recompute scores, normalize per column, P @ V
__global__ __launch_bounds__(256) void attn_pv_kernel(
        const u16* __restrict__ Qb, const u16* __restrict__ Kb,
        const u16* __restrict__ Vtb, const int* __restrict__ mask,
        const float* __restrict__ mstat, const float* __restrict__ lstat,
        u16* __restrict__ attn)
{
    __shared__ __align__(16) unsigned char ldsK[32 * 528];    // K rows (32k x 256d bf16)
    __shared__ __align__(16) unsigned char ldsV[256 * 80];    // V^T rows (256d x 32k bf16)
    __shared__ __align__(16) unsigned char ldsP[8 * 16 * 80]; // per-wave P staging

    const int tid = threadIdx.x, lane = tid & 31, w = tid >> 5;
    const int hl = lane >> 4, l15 = lane & 15;
    const int bh = blockIdx.y, b = bh >> 2, h = bh & 3;
    const int q0 = blockIdx.x * 128 + w * 16;     // this wave's 16 q-rows
    const u16* Qh = Qb + bh * (SEQ * DH);
    const u16* Kh = Kb + bh * (SEQ * DH);
    const u16* Vh = Vtb + bh * (DH * SEQ);
    unsigned char* stg = ldsP + w * (16 * 80);
    const float scale = 0.0625f;

    // resident A-operands: Q rows q0..q0+15, full d (8 chunks)
    bf16x16 qop[8];
#pragma unroll
    for (int c = 0; c < 8; ++c)
        qop[c] = load_aop(Qh + (q0 + l15) * DH + c * 32, hl);

    f32x8 o[16] = {};   // O[16q x 256d] accumulators (16 N-tiles)

    for (int kk = 0; kk < SEQ; kk += 32) {
        // async-stage K tile (32 rows x 512B, 528B padded rows)
        // and V^T tile (256 rows x 64B, 80B padded rows)
#pragma unroll
        for (int t = 0; t < 4; ++t) {
            int ch = tid + t * 256;                  // 0..1023
            int row = ch >> 5, seg = ch & 31;
            async_copy16(ldsK + row * 528 + seg * 16,
                         Kh + (kk + row) * DH + seg * 8);
        }
#pragma unroll
        for (int t = 0; t < 4; ++t) {
            int ch = tid + t * 256;
            int row = ch >> 2, seg = ch & 3;
            async_copy16(ldsV + row * 80 + seg * 16,
                         Vh + row * SEQ + kk + seg * 8);
        }
        wait_async0();
        __syncthreads();

        // two 16-column score tiles -> normalized bf16 P staged in A-layout source form
#pragma unroll
        for (int hf = 0; hf < 2; ++hf) {
            f32x8 s = {};
            const u16* krow = (const u16*)(ldsK + (hf * 16 + l15) * 528);
#pragma unroll
            for (int c = 0; c < 8; ++c)
                s = wmma_bf16(qop[c], load_bop(krow + c * 32, hl), s);

            int kcol = kk + hf * 16 + l15;
            float mv = mstat[bh * SEQ + kcol];
            float inv = 1.0f / lstat[bh * SEQ + kcol];
#pragma unroll
            for (int e = 0; e < 8; ++e) {
                int q = q0 + hl * 8 + e;
                int mk = mask[(b * SEQ + q) * SEQ + kcol];
                float sv = mk ? s[e] * scale : -1.0e9f;
                float p = __expf(sv - mv) * inv;
                *(u16*)(stg + (hl * 8 + e) * 80 + (hf * 16 + l15) * 2) = f2bf(p);
            }
        }
        asm volatile("s_wait_dscnt 0x0" ::: "memory");  // P stores -> P loads (same wave)

        bf16x16 pa = load_aop((const u16*)(stg + l15 * 80), hl);   // P as A (16q x 32k)
#pragma unroll
        for (int j = 0; j < 16; ++j) {
            const u16* vrow = (const u16*)(ldsV + (j * 16 + l15) * 80);
            o[j] = wmma_bf16(pa, load_bop(vrow, hl), o[j]);
        }
        __syncthreads();
    }

    // write attn out bf16 [B,T,DIM] (A-matrix for the final projection GEMM)
#pragma unroll
    for (int j = 0; j < 16; ++j) {
        int col = h * DH + j * 16 + l15;
#pragma unroll
        for (int e = 0; e < 8; ++e) {
            int t = q0 + hl * 8 + e;
            attn[(b * SEQ + t) * DIMN + col] = f2bf(o[j][e]);
        }
    }
}

// ---------------------------------------------------------------------------
extern "C" void kernel_launch(void* const* d_in, const int* in_sizes, int n_in,
                              void* d_out, int out_size, void* d_ws, size_t ws_size,
                              hipStream_t stream)
{
    (void)in_sizes; (void)n_in; (void)out_size; (void)ws_size;
    const float* query  = (const float*)d_in[0];
    const float* keys   = (const float*)d_in[1];
    const float* values = (const float*)d_in[2];
    const int*   mask   = (const int*)d_in[3];
    const float* Wq = (const float*)d_in[4];  const float* bq = (const float*)d_in[5];
    const float* Wk = (const float*)d_in[6];  const float* bk = (const float*)d_in[7];
    const float* Wv = (const float*)d_in[8];  const float* bv = (const float*)d_in[9];
    const float* Wo = (const float*)d_in[10]; const float* bo = (const float*)d_in[11];

    char* ws = (char*)d_ws;
    size_t off = 0;
    auto wsalloc = [&](size_t bytes) {
        char* p = ws + off;
        off += (bytes + 255) & ~(size_t)255;
        return p;
    };
    const size_t NW = (size_t)DIMN * DIMN;   // 1M weight elems
    const size_t NX = (size_t)MTOT * DIMN;   // 8.4M activation elems

    u16* Wq_b = (u16*)wsalloc(NW * 2);
    u16* Wk_b = (u16*)wsalloc(NW * 2);
    u16* Wv_b = (u16*)wsalloc(NW * 2);
    u16* Wo_b = (u16*)wsalloc(NW * 2);
    u16* Xq_b = (u16*)wsalloc(NX * 2);
    u16* Xk_b = (u16*)wsalloc(NX * 2);
    u16* Xv_b = (u16*)wsalloc(NX * 2);
    u16* Qh_b = (u16*)wsalloc(NX * 2);       // [B,H,T,DH]
    u16* Kh_b = (u16*)wsalloc(NX * 2);       // [B,H,T,DH]
    u16* Vt_b = (u16*)wsalloc(NX * 2);       // [B,H,DH,T]
    u16* At_b = (u16*)wsalloc(NX * 2);       // attn out [B,T,DIM]
    float* mstat = (float*)wsalloc((size_t)BATCH * HEADS * SEQ * 4);
    float* lstat = (float*)wsalloc((size_t)BATCH * HEADS * SEQ * 4);

    // f32 -> bf16 conversions
    cvt_kernel<<<(int)((NW + 255) / 256), 256, 0, stream>>>(Wq, Wq_b, (int)NW);
    cvt_kernel<<<(int)((NW + 255) / 256), 256, 0, stream>>>(Wk, Wk_b, (int)NW);
    cvt_kernel<<<(int)((NW + 255) / 256), 256, 0, stream>>>(Wv, Wv_b, (int)NW);
    cvt_kernel<<<(int)((NW + 255) / 256), 256, 0, stream>>>(Wo, Wo_b, (int)NW);
    cvt_kernel<<<(int)((NX + 255) / 256), 256, 0, stream>>>(query,  Xq_b, (int)NX);
    cvt_kernel<<<(int)((NX + 255) / 256), 256, 0, stream>>>(keys,   Xk_b, (int)NX);
    cvt_kernel<<<(int)((NX + 255) / 256), 256, 0, stream>>>(values, Xv_b, (int)NX);

    // projections
    dim3 ggrid(MTOT / 128, DIMN / 128);
    gemm_bf16_kernel<0><<<ggrid, 256, 0, stream>>>(Xq_b, Wq_b, bq, Qh_b);
    gemm_bf16_kernel<0><<<ggrid, 256, 0, stream>>>(Xk_b, Wk_b, bk, Kh_b);
    gemm_bf16_kernel<1><<<ggrid, 256, 0, stream>>>(Xv_b, Wv_b, bv, Vt_b);

    // attention (column softmax over q, per-k stats)
    dim3 agrid(SEQ / 128, BATCH * HEADS);
    attn_stats_kernel<<<agrid, 256, 0, stream>>>(Qh_b, Kh_b, mask, mstat, lstat);
    attn_pv_kernel<<<agrid, 256, 0, stream>>>(Qh_b, Kh_b, Vt_b, mask, mstat, lstat, At_b);

    // output projection -> f32 d_out
    gemm_bf16_kernel<2><<<ggrid, 256, 0, stream>>>(At_b, Wo_b, bo, (float*)d_out);
}
